// DistributedDotGAT_85220741087570
// MI455X (gfx1250) — matrix operand order repaired and checked by reading, not compiled, not run on hardware
//
#include <hip/hip_runtime.h>

// ---------------------------------------------------------------------------
// CDNA5 (gfx1250) WMMA implementation of DistributedDotGAT forward.
// All dense math via v_wmma_f32_16x16x32_bf16 (fp32 accumulate).
// GEMM convention: C = alpha * A(MxK, row-major) @ B^T where B is stored
// [N][K] row-major ("BT layout"). Weights are transposed once at convert time.
// Tile staging uses GLOBAL_LOAD_ASYNC_TO_LDS_B128 (ASYNCcnt) with LDS
// double buffering. All M,N multiples of 64, K multiples of 32 -> no guards.
// ---------------------------------------------------------------------------

typedef __attribute__((ext_vector_type(16))) __bf16          v16bf;
typedef __attribute__((ext_vector_type(8)))  float           v8f;
typedef __attribute__((ext_vector_type(8)))  unsigned short  v8us;

union FragBF { v8us h[2]; unsigned short u[16]; v16bf v; };

constexpr int kB  = 8;
constexpr int kA  = 512;
constexpr int kIN = 1024;
constexpr int kHD = 512;
constexpr int kNH = 8;
constexpr int kN  = 64;
constexpr int kR  = 32;
constexpr int kSTEPS = 3;
constexpr int kT  = kB * kA;          // 4096 tokens

#define DEVFN __device__ __forceinline__

DEVFN unsigned short f32_to_bf16(float f) {
  unsigned int x = __float_as_uint(f);
  return (unsigned short)((x + 0x7FFFu + ((x >> 16) & 1u)) >> 16);
}

// --------------------------- fp32 -> bf16 (RNE) ----------------------------
__global__ void f2bf_kernel(const float* __restrict__ in,
                            unsigned short* __restrict__ out, long n) {
  long i = (long)blockIdx.x * blockDim.x + threadIdx.x;
  long stride = (long)gridDim.x * blockDim.x;
  for (; i < n; i += stride) out[i] = f32_to_bf16(in[i]);
}

// ------------------ fp32 -> bf16 with transpose (batched) ------------------
// in : [rows][cols] fp32 ; out: [cols][rows] bf16 (batch stride rows*cols)
__global__ __launch_bounds__(256)
void f2bf_t_kernel(const float* __restrict__ in,
                   unsigned short* __restrict__ out, int rows, int cols) {
  __shared__ unsigned short tile[32][33];
  const long bofs = (long)blockIdx.z * rows * cols;
  in  += bofs;
  out += bofs;
  const int r0 = blockIdx.y * 32, c0 = blockIdx.x * 32;
  const int tc = threadIdx.x & 31, tr8 = threadIdx.x >> 5;   // 0..7
#pragma unroll
  for (int i = 0; i < 4; ++i) {
    int r = tr8 + i * 8;
    tile[r][tc] = f32_to_bf16(in[(long)(r0 + r) * cols + (c0 + tc)]);
  }
  __syncthreads();
#pragma unroll
  for (int i = 0; i < 4; ++i) {
    int c = tr8 + i * 8;
    out[(long)(c0 + c) * rows + (r0 + tc)] = tile[tc][c];
  }
}

// --------------------------- WMMA tiled GEMM (BT) --------------------------
// 128 threads = 4 waves. Block tile 64x64, K chunk 32, LDS double-buffered.
// Wave w computes rows [16w,16w+16) x all 64 cols: 4 WMMAs/chunk, shared A frag.
// LDS rows are 80B-stride (16B aligned, conflict-free b128 fragment loads).
// Staging: per-thread GLOBAL_LOAD_ASYNC_TO_LDS_B128 into buffer p^1 while
// computing from buffer p; async loads complete in order, so waiting
// ASYNCcnt<=4 guarantees the previous panel's 4 copies have landed.
__global__ __launch_bounds__(128)
void gemm_bt_kernel(const unsigned short* __restrict__ A,
                    const unsigned short* __restrict__ Bm,  // [N][K]
                    float* __restrict__ C,
                    int M, int N, int K,
                    long sA, long sB, long sC, float alpha) {
  __shared__ alignas(16) unsigned short As[2][64][40];  // 5120 B per buffer
  __shared__ alignas(16) unsigned short Bs[2][64][40];
  constexpr unsigned kBufStride = 64 * 40 * 2;          // 5120 B
  constexpr unsigned kRowStride = 40 * 2;               // 80 B

  const int batch = blockIdx.z;
  A  += (long)batch * sA;
  Bm += (long)batch * sB;
  C  += (long)batch * sC;

  const int tid  = threadIdx.x;
  const int lane = tid & 31;
  const int wave = tid >> 5;

  const int m0 = blockIdx.y * 64;
  const int n0 = blockIdx.x * 64;

  const int fr = lane & 15;
  const int hi = lane >> 4;
  const int wm = wave * 16;

  // staging map: thread t copies rows (t>>2) and (t>>2)+32, kseg = (t&3)*8
  const int srow = tid >> 2;
  const int kseg = (tid & 3) * 8;

  // per-thread global sources (advance 32 elems / 64B per chunk)
  const unsigned short* ga0 = &A[(long)(m0 + srow) * K + kseg];
  const unsigned short* ga1 = &A[(long)(m0 + srow + 32) * K + kseg];
  const unsigned short* gb0 = &Bm[(long)(n0 + srow) * K + kseg];
  const unsigned short* gb1 = &Bm[(long)(n0 + srow + 32) * K + kseg];

  // per-thread LDS destinations (buffer 0); buffer 1 = +kBufStride
  const unsigned la0 = (unsigned)(unsigned long long)(void*)&As[0][srow][kseg];
  const unsigned la1 = la0 + 32 * kRowStride;
  const unsigned lb0 = (unsigned)(unsigned long long)(void*)&Bs[0][srow][kseg];
  const unsigned lb1 = lb0 + 32 * kRowStride;

  auto issue_async = [&](int p, int k0) {
    const unsigned o = p ? kBufStride : 0u;
    asm volatile(
        "global_load_async_to_lds_b128 %0, %4, off\n\t"
        "global_load_async_to_lds_b128 %1, %5, off\n\t"
        "global_load_async_to_lds_b128 %2, %6, off\n\t"
        "global_load_async_to_lds_b128 %3, %7, off"
        :
        : "v"(la0 + o), "v"(la1 + o), "v"(lb0 + o), "v"(lb1 + o),
          "v"(ga0 + k0), "v"(ga1 + k0), "v"(gb0 + k0), "v"(gb1 + k0)
        : "memory");
  };

  issue_async(0, 0);

  v8f acc0 = {}, acc1 = {}, acc2 = {}, acc3 = {};

  for (int k0 = 0; k0 < K; k0 += 32) {
    const int  p    = (k0 >> 5) & 1;
    const bool more = (k0 + 32) < K;   // uniform across block

    if (more) {
      issue_async(p ^ 1, k0 + 32);
      asm volatile("s_wait_asynccnt 0x4" ::: "memory");
    } else {
      asm volatile("s_wait_asynccnt 0x0" ::: "memory");
    }
    __syncthreads();   // buffer p fully populated (all waves)

    // ---- A fragment: row wm+fr, 16 contiguous K at 16*hi ----
    FragBF fa;
    fa.h[0] = *reinterpret_cast<const v8us*>(&As[p][wm + fr][hi * 16]);
    fa.h[1] = *reinterpret_cast<const v8us*>(&As[p][wm + fr][hi * 16 + 8]);

    // ---- B fragments: col fr, 16 contiguous K at 16*hi ----
    FragBF fb0, fb1, fb2, fb3;
    fb0.h[0] = *reinterpret_cast<const v8us*>(&Bs[p][fr][hi * 16]);
    fb0.h[1] = *reinterpret_cast<const v8us*>(&Bs[p][fr][hi * 16 + 8]);
    fb1.h[0] = *reinterpret_cast<const v8us*>(&Bs[p][16 + fr][hi * 16]);
    fb1.h[1] = *reinterpret_cast<const v8us*>(&Bs[p][16 + fr][hi * 16 + 8]);
    fb2.h[0] = *reinterpret_cast<const v8us*>(&Bs[p][32 + fr][hi * 16]);
    fb2.h[1] = *reinterpret_cast<const v8us*>(&Bs[p][32 + fr][hi * 16 + 8]);
    fb3.h[0] = *reinterpret_cast<const v8us*>(&Bs[p][48 + fr][hi * 16]);
    fb3.h[1] = *reinterpret_cast<const v8us*>(&Bs[p][48 + fr][hi * 16 + 8]);

    acc0 = __builtin_amdgcn_wmma_f32_16x16x32_bf16(false, fa.v, false, fb0.v,
                                                   (short)0, acc0, false, false);
    acc1 = __builtin_amdgcn_wmma_f32_16x16x32_bf16(false, fa.v, false, fb1.v,
                                                   (short)0, acc1, false, false);
    acc2 = __builtin_amdgcn_wmma_f32_16x16x32_bf16(false, fa.v, false, fb2.v,
                                                   (short)0, acc2, false, false);
    acc3 = __builtin_amdgcn_wmma_f32_16x16x32_bf16(false, fa.v, false, fb3.v,
                                                   (short)0, acc3, false, false);
    __syncthreads();   // readers of buffer p done before it is re-filled
  }

  // ---- store: VGPR e -> row e + 8*hi, col = fr (+16 per tile) ----
#pragma unroll
  for (int e = 0; e < 8; ++e) {
    const long r = m0 + wm + e + hi * 8;
    float* cr = C + r * N + n0 + fr;
    cr[0]  = acc0[e] * alpha;
    cr[16] = acc1[e] * alpha;
    cr[32] = acc2[e] * alpha;
    cr[48] = acc3[e] * alpha;
  }
}

// --------------------------- block reductions ------------------------------
DEVFN float block_reduce_sum(float v, float* sm) {
  const int tid = threadIdx.x;
  sm[tid] = v;
  __syncthreads();
  for (int s = 128; s > 0; s >>= 1) {
    if (tid < s) sm[tid] += sm[tid + s];
    __syncthreads();
  }
  float r = sm[0];
  __syncthreads();
  return r;
}

DEVFN float block_reduce_max(float v, float* sm) {
  const int tid = threadIdx.x;
  sm[tid] = v;
  __syncthreads();
  for (int s = 128; s > 0; s >>= 1) {
    if (tid < s) sm[tid] = fmaxf(sm[tid], sm[tid + s]);
    __syncthreads();
  }
  float r = sm[0];
  __syncthreads();
  return r;
}

// ---------------- fused (bias +) optional-swish + LayerNorm ----------------
template<bool SWISH>
__global__ __launch_bounds__(256)
void act_ln_kernel(const float* __restrict__ in, const float* __restrict__ bias,
                   const float* __restrict__ g, const float* __restrict__ b,
                   float* __restrict__ out, int cols) {
  __shared__ float sm[256];
  const int row = blockIdx.x;
  const float* x = in + (long)row * cols;
  float* y = out + (long)row * cols;

  float vals[4];
  int nv = 0;
  float s = 0.f;
  for (int c = threadIdx.x; c < cols; c += 256) {
    float v = x[c];
    if (bias) v += bias[c];
    if (SWISH) v = v / (1.f + __expf(-v));   // x * sigmoid(x)
    vals[nv++] = v;
    s += v;
  }
  const float mean = block_reduce_sum(s, sm) / (float)cols;
  float s2 = 0.f;
  for (int i = 0; i < nv; ++i) { float d = vals[i] - mean; s2 += d * d; }
  const float var = block_reduce_sum(s2, sm) / (float)cols;
  const float inv = rsqrtf(var + 1e-5f);
  nv = 0;
  for (int c = threadIdx.x; c < cols; c += 256)
    y[c] = (vals[nv++] - mean) * inv * g[c] + b[c];
}

// ------------------------ masked row softmax -------------------------------
__global__ __launch_bounds__(256)
void softmax_kernel(float* __restrict__ scores, const float* __restrict__ conn,
                    int Adim) {
  __shared__ float sm[256];
  const int row = blockIdx.x;           // b*A + a
  const int a   = row % Adim;
  float* x = scores + (long)row * Adim;
  const float* cr = conn + (long)a * Adim;

  float vals[4];
  int nv = 0;
  float mx = -3.4e38f;
  for (int c = threadIdx.x; c < Adim; c += 256) {
    float v = x[c] + (cr[c] == 0.f ? -1e9f : 0.f);
    vals[nv++] = v;
    mx = fmaxf(mx, v);
  }
  mx = block_reduce_max(mx, sm);
  float sum = 0.f;
  for (int i = 0; i < nv; ++i) { vals[i] = __expf(vals[i] - mx); sum += vals[i]; }
  sum = block_reduce_sum(sum, sm);
  const float inv = 1.f / sum;
  nv = 0;
  for (int c = threadIdx.x; c < Adim; c += 256) x[c] = vals[nv++] * inv;
}

// ---------------- per-element winning-head (max |value|) -------------------
__global__ void head_select_kernel(const float* __restrict__ stk,
                                   float* __restrict__ h, long n, long hs,
                                   int nh) {
  long i = (long)blockIdx.x * blockDim.x + threadIdx.x;
  long stride = (long)gridDim.x * blockDim.x;
  for (; i < n; i += stride) {
    float best = stk[i];
    float ab = fabsf(best);
    for (int hh = 1; hh < nh; ++hh) {
      float v = stk[(long)hh * hs + i];
      float av = fabsf(v);
      if (av > ab) { ab = av; best = v; }   // first head wins ties (argmax)
    }
    h[i] = best;
  }
}

// ---------------------------------------------------------------------------
static inline int cdiv(int a, int b) { return (a + b - 1) / b; }

static void launch_gemm(hipStream_t s, const unsigned short* A,
                        const unsigned short* B, float* C, int M, int N, int K,
                        int batches, long sA, long sB, long sC, float alpha) {
  dim3 grid(cdiv(N, 64), cdiv(M, 64), batches);
  gemm_bt_kernel<<<grid, 128, 0, s>>>(A, B, C, M, N, K, sA, sB, sC, alpha);
}

extern "C" void kernel_launch(void* const* d_in, const int* in_sizes, int n_in,
                              void* d_out, int out_size, void* d_ws,
                              size_t ws_size, hipStream_t stream) {
  // setup_inputs() dict order: x, params{...}, connectivity
  const float* x    = (const float*)d_in[0];
  const float* inW1 = (const float*)d_in[1];
  const float* ing1 = (const float*)d_in[2];
  const float* inb1 = (const float*)d_in[3];
  const float* inW2 = (const float*)d_in[4];
  const float* ing2 = (const float*)d_in[5];
  const float* inb2 = (const float*)d_in[6];
  const float* inW3 = (const float*)d_in[7];
  const float* Wq   = (const float*)d_in[8];
  const float* Wk   = (const float*)d_in[9];
  const float* Wv   = (const float*)d_in[10];
  const float* g1   = (const float*)d_in[11];
  const float* b1   = (const float*)d_in[12];
  const float* W1   = (const float*)d_in[13];
  const float* bb1  = (const float*)d_in[14];
  const float* g2   = (const float*)d_in[15];
  const float* b2   = (const float*)d_in[16];
  const float* W2   = (const float*)d_in[17];
  const float* bb2  = (const float*)d_in[18];
  const float* go   = (const float*)d_in[19];
  const float* bo   = (const float*)d_in[20];
  const float* WU   = (const float*)d_in[21];
  const float* WVp  = (const float*)d_in[22];
  const float* conn = (const float*)d_in[23];
  float* out = (float*)d_out;

  // ---- workspace bump allocator (256B aligned) ----
  char* wp = (char*)d_ws;
  auto alloc = [&](size_t bytes) -> void* {
    void* r = (void*)wp;
    wp += (bytes + 255) & ~(size_t)255;
    return r;
  };
  auto allocU = [&](long e) { return (unsigned short*)alloc((size_t)e * 2); };
  auto allocF = [&](long e) { return (float*)alloc((size_t)e * 4); };

  const long HH = (long)kHD * kHD;          // 262144
  // bf16 weights, stored transposed [N][K]
  unsigned short* wi1   = allocU((long)kIN * 2 * kHD);
  unsigned short* wi2   = allocU((long)2 * kHD * 2 * kHD);
  unsigned short* wi3   = allocU((long)2 * kHD * kHD);
  unsigned short* wq_bf = allocU((long)kNH * HH);
  unsigned short* wk_bf = allocU((long)kNH * HH);
  unsigned short* wv_bf = allocU((long)kNH * HH);
  unsigned short* w1_bf = allocU((long)kNH * HH);
  unsigned short* w2_bf = allocU((long)kNH * HH);
  unsigned short* wu_bf = allocU((long)kHD * kN * kR);
  unsigned short* wo_bf = allocU((long)kHD * kN * kR);
  // bf16 activations
  unsigned short* x_bf  = allocU((long)kT * kIN);
  unsigned short* a1_bf = allocU((long)kT * kIN);
  unsigned short* h_bf  = allocU((long)kT * kHD);
  unsigned short* q_bf  = allocU((long)kT * kHD);
  unsigned short* k_bf  = allocU((long)kT * kHD);
  unsigned short* vT_bf = allocU((long)kT * kHD);   // per-batch transposed V
  unsigned short* p_bf  = allocU((long)kB * kA * kA);
  unsigned short* t_bf  = allocU((long)kT * kHD);
  unsigned short* u_bf  = allocU((long)kT * kN * kR);
  unsigned short* vm_bf = allocU((long)kT * kN * kR);
  // fp32 temporaries
  float* tmpf  = allocF((long)kT * kN * kR);    // largest fp32 GEMM output
  float* tmpf2 = allocF((long)kT * kIN);        // LN outputs (<=1024 cols)
  float* hf    = allocF((long)kT * kHD);
  float* stk   = allocF((long)kNH * kT * kHD);  // stacked head outputs
  float* scor  = allocF((long)kB * kA * kA);    // one head's scores

  auto f2bf = [&](const float* src, unsigned short* dst, long n) {
    f2bf_kernel<<<dim3(2048), dim3(256), 0, stream>>>(src, dst, n);
  };
  auto f2bf_t = [&](const float* src, unsigned short* dst, int rows, int cols,
                    int batches) {
    dim3 grid(cols / 32, rows / 32, batches);
    f2bf_t_kernel<<<grid, 256, 0, stream>>>(src, dst, rows, cols);
  };
  auto swish_ln = [&](const float* in, const float* bias, const float* g,
                      const float* b, float* o, int cols) {
    act_ln_kernel<true><<<dim3(kT), dim3(256), 0, stream>>>(in, bias, g, b, o, cols);
  };
  auto plain_ln = [&](const float* in, const float* bias, const float* g,
                      const float* b, float* o, int cols) {
    act_ln_kernel<false><<<dim3(kT), dim3(256), 0, stream>>>(in, bias, g, b, o, cols);
  };

  const float invScale = 1.f / sqrtf((float)kHD);

  // ---- convert weights (transposed to [N][K]) and input ----
  f2bf(x, x_bf, (long)kT * kIN);
  f2bf_t(inW1, wi1, kIN, 2 * kHD, 1);
  f2bf_t(inW2, wi2, 2 * kHD, 2 * kHD, 1);
  f2bf_t(inW3, wi3, 2 * kHD, kHD, 1);
  f2bf_t(Wq, wq_bf, kHD, kHD, kNH);
  f2bf_t(Wk, wk_bf, kHD, kHD, kNH);
  f2bf_t(Wv, wv_bf, kHD, kHD, kNH);
  f2bf_t(W1, w1_bf, kHD, kHD, kNH);
  f2bf_t(W2, w2_bf, kHD, kHD, kNH);
  f2bf_t(WU, wu_bf, kHD, kN * kR, 1);
  f2bf_t(WVp, wo_bf, kHD, kN * kR, 1);

  // ---- input projection MLP ----
  launch_gemm(stream, x_bf, wi1, tmpf, kT, 2 * kHD, kIN, 1, 0, 0, 0, 1.f);
  swish_ln(tmpf, nullptr, ing1, inb1, tmpf2, 2 * kHD);
  f2bf(tmpf2, a1_bf, (long)kT * 2 * kHD);
  launch_gemm(stream, a1_bf, wi2, tmpf, kT, 2 * kHD, 2 * kHD, 1, 0, 0, 0, 1.f);
  swish_ln(tmpf, nullptr, ing2, inb2, tmpf2, 2 * kHD);
  f2bf(tmpf2, a1_bf, (long)kT * 2 * kHD);
  launch_gemm(stream, a1_bf, wi3, hf, kT, kHD, 2 * kHD, 1, 0, 0, 0, 1.f);
  f2bf(hf, h_bf, (long)kT * kHD);

  // ---- message-passing steps ----
  const long sAA = (long)kA * kA;    // per-batch stride for scores
  const long sAH = (long)kA * kHD;   // per-batch stride for Q/K/V
  for (int step = 0; step < kSTEPS; ++step) {
    for (int hh = 0; hh < kNH; ++hh) {
      const unsigned short* wqh = wq_bf + (long)hh * HH;
      const unsigned short* wkh = wk_bf + (long)hh * HH;
      const unsigned short* wvh = wv_bf + (long)hh * HH;
      const unsigned short* w1h = w1_bf + (long)hh * HH;
      const unsigned short* w2h = w2_bf + (long)hh * HH;

      launch_gemm(stream, h_bf, wqh, tmpf, kT, kHD, kHD, 1, 0, 0, 0, 1.f);
      f2bf(tmpf, q_bf, (long)kT * kHD);
      launch_gemm(stream, h_bf, wkh, tmpf, kT, kHD, kHD, 1, 0, 0, 0, 1.f);
      f2bf(tmpf, k_bf, (long)kT * kHD);
      launch_gemm(stream, h_bf, wvh, tmpf, kT, kHD, kHD, 1, 0, 0, 0, 1.f);
      // V -> per-batch transposed bf16 [b][HD][A] (BT layout for alpha @ V)
      f2bf_t(tmpf, vT_bf, kA, kHD, kB);

      // scores = Q K^T / sqrt(HD); K is naturally [N=A][K=HD]
      launch_gemm(stream, q_bf, k_bf, scor, kA, kA, kHD, kB, sAH, sAH, sAA,
                  invScale);
      softmax_kernel<<<dim3(kT), dim3(256), 0, stream>>>(scor, conn, kA);
      f2bf(scor, p_bf, (long)kB * kA * kA);

      // Hv = alpha @ V  (B operand = V^T in BT layout)
      launch_gemm(stream, p_bf, vT_bf, tmpf, kA, kHD, kA, kB, sAA, sAH, sAH, 1.f);

      swish_ln(tmpf, nullptr, g1 + (long)hh * kHD, b1 + (long)hh * kHD, tmpf2, kHD);
      f2bf(tmpf2, t_bf, (long)kT * kHD);
      launch_gemm(stream, t_bf, w1h, tmpf, kT, kHD, kHD, 1, 0, 0, 0, 1.f);
      swish_ln(tmpf, bb1 + (long)hh * kHD, g2 + (long)hh * kHD,
               b2 + (long)hh * kHD, tmpf2, kHD);
      f2bf(tmpf2, t_bf, (long)kT * kHD);
      launch_gemm(stream, t_bf, w2h, tmpf, kT, kHD, kHD, 1, 0, 0, 0, 1.f);
      plain_ln(tmpf, bb2 + (long)hh * kHD, go + (long)hh * kHD,
               bo + (long)hh * kHD, stk + (long)hh * kT * kHD, kHD);
    }
    head_select_kernel<<<dim3(4096), dim3(256), 0, stream>>>(
        stk, hf, (long)kT * kHD, (long)kT * kHD, kNH);
    f2bf(hf, h_bf, (long)kT * kHD);
  }

  // ---- low-rank output: U = hWU, V = hWV, recon = U V^T / R ----
  launch_gemm(stream, h_bf, wu_bf, tmpf, kT, kN * kR, kHD, 1, 0, 0, 0, 1.f);
  f2bf(tmpf, u_bf, (long)kT * kN * kR);
  launch_gemm(stream, h_bf, wo_bf, tmpf, kT, kN * kR, kHD, 1, 0, 0, 0, 1.f);
  f2bf(tmpf, vm_bf, (long)kT * kN * kR);

  // per-token 64x64 = U(64x32) @ V(64x32)^T; V rows are naturally [N=m][K=r]
  launch_gemm(stream, u_bf, vm_bf, out, kN, kN, kR, kT,
              (long)kN * kR, (long)kN * kR, (long)kN * kN, 1.f / (float)kR);
}
